// Graphormer_48455821033932
// MI455X (gfx1250) — matrix operand order repaired and checked by reading, not compiled, not run on hardware
//
#include <hip/hip_runtime.h>
#include <hip/hip_bf16.h>

typedef __attribute__((ext_vector_type(16))) _Float16 v16h;
typedef __attribute__((ext_vector_type(8)))  float    v8f;

constexpr int kB = 256, kN = 128, kND = 64, kH = 128, kNH = 8, kDH = 16;
constexpr int kFF = 512, kNL = 4, kEPG = 256, kCOND = 16;
constexpr int kMAXD = 10, kMAXDEG = 10, kMLPH = 256, kL = kN + 1;

#define USE_ASYNC_LDS 1

// ---------------- WMMA helpers (CDNA5 16x16x32 f16 -> f32) ----------------

__device__ inline v8f wmma32(v16h a, v16h b, v8f c) {
  return __builtin_amdgcn_wmma_f32_16x16x32_f16(false, a, false, b, (short)0, c,
                                                false, false);
}

// Packed fragment layout: 512 f16 per 16x32 tile, element (lane, i) holds
//   A view: row = lane&15, d = i + 16*(lane>>4)
//   B view: col = lane&15, krel = (lane>>4)*16 + i
// so each lane's 16 values are contiguous (32 B) -> b128 loads.
__device__ inline v16h fragP(const _Float16* P, int lane) {
  return *(const v16h*)(P + (lane << 4));
}

// inverse map: packed index of A element (row rr in [0,16), k-offset d in [0,32))
__device__ inline int packA_idx(int rr, int d) {
  int hf = (d >> 4) & 1;
  int i  = d & 15;
  return ((rr + 16 * hf) << 4) | i;
}

// ---------------- async global -> LDS tile staging ----------------

__device__ inline void stage_tile(_Float16* lds, const _Float16* g, int n16,
                                  int t, int nthreads) {
#if USE_ASYNC_LDS
  unsigned lbase = (unsigned)(size_t)lds;
  for (int idx = t; idx < n16; idx += nthreads) {
    unsigned long long ga = (unsigned long long)(size_t)g + (unsigned long long)idx * 16ull;
    asm volatile("global_load_async_to_lds_b128 %0, %1, off"
                 :: "v"(lbase + (unsigned)idx * 16u), "v"(ga)
                 : "memory");
  }
  asm volatile("s_wait_asynccnt 0x0" ::: "memory");
#else
  for (int idx = t; idx < n16; idx += nthreads)
    ((v16h*)lds)[idx] = ((const v16h*)g)[idx];
#endif
}

// ---------------- weight pre-pack (f32 row-major KxN -> packed f16 frags) ----

__global__ void k_pack(const float* W, _Float16* P, int K, int Nc) {
  int idx = blockIdx.x * 256 + threadIdx.x;
  int nKt = K >> 5;
  int total = nKt * (Nc >> 4) * 512;
  if (idx >= total) return;
  int i = idx & 15, lane = (idx >> 4) & 31;
  int frag = idx >> 9;
  int kc = frag % nKt, nt = frag / nKt;
  int c = lane & 15, hf = lane >> 4;
  int krel = hf * 16 + i;
  P[idx] = (_Float16)W[(size_t)(kc * 32 + krel) * Nc + nt * 16 + c];
}

// ---------------- degrees (block = batch, 128 threads) ----------------

__global__ void k_deg(const int* edge, int* din, int* dout) {
  int b = blockIdx.x, t = threadIdx.x;
  __shared__ int sdi[kN], sdo[kN];
  sdi[t] = 0; sdo[t] = 0;
  __syncthreads();
  const int* src = edge + (size_t)b * 2 * kEPG;
  const int* tgt = src + kEPG;
  for (int e = t; e < kEPG; e += 128) {
    atomicAdd(&sdo[src[e]], 1);
    atomicAdd(&sdi[tgt[e]], 1);
  }
  __syncthreads();
  din[b * kN + t]  = min(sdi[t], kMAXDEG);
  dout[b * kN + t] = min(sdo[t], kMAXDEG);
}

// ---------------- shortest paths via bitmask frontiers ----------------

__global__ void k_sp(const int* edge, unsigned char* sp) {
  int b = blockIdx.x, t = threadIdx.x;
  __shared__ unsigned adj[kN][4];
  adj[t][0] = adj[t][1] = adj[t][2] = adj[t][3] = 0u;
  __syncthreads();
  const int* src = edge + (size_t)b * 2 * kEPG;
  const int* tgt = src + kEPG;
  for (int e = t; e < kEPG; e += 128) {
    int s = src[e], tg = tgt[e];
    atomicOr(&adj[s][tg >> 5], 1u << (tg & 31));
  }
  __syncthreads();
  unsigned char* row = sp + (size_t)b * kL * kL + (size_t)(t + 1) * kL;
  row[0] = 1;
  for (int j = 0; j < kN; ++j) row[j + 1] = (j == t) ? 0 : (unsigned char)(kMAXD + 1);
  if (t == 0) {
    unsigned char* r0 = sp + (size_t)b * kL * kL;
    r0[0] = 0;
    for (int j = 1; j < kL; ++j) r0[j] = 1;
  }
  unsigned cur[4] = {adj[t][0], adj[t][1], adj[t][2], adj[t][3]};
  unsigned vis[4] = {0, 0, 0, 0};
  vis[t >> 5] |= 1u << (t & 31);
  for (int d = 1; d <= kMAXD; ++d) {
#pragma unroll
    for (int w = 0; w < 4; ++w) {
      unsigned nw = cur[w] & ~vis[w];
      while (nw) {
        int bp = __builtin_ctz(nw);
        nw &= nw - 1;
        row[w * 32 + bp + 1] = (unsigned char)d;
        vis[w] |= 1u << bp;
      }
    }
    unsigned nx[4] = {0, 0, 0, 0};
#pragma unroll
    for (int w = 0; w < 4; ++w) {
      unsigned m = cur[w];
      while (m) {
        int j = w * 32 + __builtin_ctz(m);
        m &= m - 1;
        nx[0] |= adj[j][0]; nx[1] |= adj[j][1];
        nx[2] |= adj[j][2]; nx[3] |= adj[j][3];
      }
    }
    cur[0] = nx[0]; cur[1] = nx[1]; cur[2] = nx[2]; cur[3] = nx[3];
  }
}

// ---------------- embedding: xd = [cls; x@Wi + bi + cent] ----------------

__global__ void k_embed(const float* x, const float* cond, const _Float16* WiP,
                        const float* bi, const float* in_emb, const float* out_emb,
                        const int* din, const int* dout, const float* Wcls,
                        const float* bcls, float* xd) {
  int b = blockIdx.x, t = threadIdx.x;  // 128 threads = 4 waves
  __shared__ __align__(32) _Float16 aX[8][2][512];  // packed A tiles, 16 KB
  const float* xb = x + (size_t)b * kN * kND;
  for (int idx = t; idx < 8192; idx += 128) {
    int i = idx & 15, lane2 = (idx >> 4) & 31, kc = (idx >> 9) & 1, mt2 = idx >> 10;
    int d = i + 16 * (lane2 >> 4);
    int row = mt2 * 16 + (lane2 & 15);
    aX[mt2][kc][((lane2) << 4) | i] = (_Float16)xb[row * kND + kc * 32 + d];
  }
  __syncthreads();
  int wave = t >> 5, lane = t & 31;
  float* xdb = xd + (size_t)b * kL * kH;
  for (int mt = wave; mt < 8; mt += 4) {
    v16h afr[2];
#pragma unroll
    for (int kc = 0; kc < 2; ++kc) afr[kc] = fragP(&aX[mt][kc][0], lane);
    for (int nt = 0; nt < 8; ++nt) {
      v8f acc = {};
#pragma unroll
      for (int kc = 0; kc < 2; ++kc)
        acc = wmma32(afr[kc], fragP(WiP + (((size_t)nt * 2 + kc) << 9), lane), acc);
      int hf = lane >> 4, n = nt * 16 + (lane & 15);
#pragma unroll
      for (int v = 0; v < 8; ++v) {
        int m = mt * 16 + v + 8 * hf;  // node index
        float val = acc[v] + bi[n] + in_emb[din[b * kN + m] * kH + n] +
                    out_emb[dout[b * kN + m] * kH + n];
        xdb[(size_t)(m + 1) * kH + n] = val;
      }
    }
  }
  if (t < kH) {  // CLS token
    float s = bcls[t];
    const float* cb = cond + (size_t)b * kCOND;
    for (int c = 0; c < kCOND; ++c) s += cb[c] * Wcls[c * kH + t];
    xdb[t] = s;
  }
}

// ------ LayerNorm of a 16-row strip, written to LDS in packed-A order ------

__device__ inline void ln_strip_packed(const float* xb, int mt, const float* g,
                                       const float* be, _Float16 (*aP)[512], int t) {
  if (t < 16) {
    int r = mt * 16 + t;
    if (r < kL) {
      const float* xr = xb + (size_t)r * kH;
      float m = 0.f;
      for (int i = 0; i < kH; ++i) m += xr[i];
      m *= (1.f / kH);
      float v = 0.f;
      for (int i = 0; i < kH; ++i) { float d0 = xr[i] - m; v += d0 * d0; }
      v *= (1.f / kH);
      float is = rsqrtf(v + 1e-5f);
      for (int ci = 0; ci < kH; ++ci) {
        float val = (xr[ci] - m) * is * g[ci] + be[ci];
        aP[ci >> 5][packA_idx(t, ci & 31)] = (_Float16)val;
      }
    } else {
      for (int ci = 0; ci < kH; ++ci)
        aP[ci >> 5][packA_idx(t, ci & 31)] = (_Float16)0.f;
    }
  }
}

// ---------------- LN1 + QKV projections ----------------

__global__ void k_qkv(const float* xd, const float* g, const float* be,
                      const _Float16* WqP, const float* bq, const _Float16* WkP,
                      const float* bk, const _Float16* WvP, const float* bv,
                      _Float16* qh, _Float16* kh, _Float16* vh) {
  int b = blockIdx.x, mt = blockIdx.y, t = threadIdx.x;  // 128 threads
  __shared__ __align__(32) _Float16 aP[4][512];
  ln_strip_packed(xd + (size_t)b * kL * kH, mt, g, be, aP, t);
  __syncthreads();
  int wave = t >> 5, lane = t & 31;
  v16h afr[4];
#pragma unroll
  for (int kc = 0; kc < 4; ++kc) afr[kc] = fragP(&aP[kc][0], lane);
  for (int tile = wave; tile < 24; tile += 4) {
    int mat = tile >> 3, nt = tile & 7;
    const _Float16* WP = (mat == 0) ? WqP : (mat == 1 ? WkP : WvP);
    const float* bias  = (mat == 0) ? bq : (mat == 1 ? bk : bv);
    _Float16* outp     = (mat == 0) ? qh : (mat == 1 ? kh : vh);
    v8f acc = {};
#pragma unroll
    for (int kc = 0; kc < 4; ++kc)
      acc = wmma32(afr[kc], fragP(WP + (((size_t)nt * 4 + kc) << 9), lane), acc);
    int hf = lane >> 4, n = nt * 16 + (lane & 15);
    _Float16* ob = outp + (size_t)b * kL * kH;
#pragma unroll
    for (int v = 0; v < 8; ++v) {
      int r = mt * 16 + v + 8 * hf;
      if (r < kL) ob[(size_t)r * kH + n] = (_Float16)(acc[v] + bias[n]);
    }
  }
}

// ---------------- attention (one wave per (b, head, q-strip)) ----------------

__global__ void k_attn(const _Float16* qh, const _Float16* kh, const _Float16* vh,
                       const unsigned char* sp, const float* sp_bias, _Float16* aoP) {
  int b = blockIdx.x, h = blockIdx.y, mt = blockIdx.z;
  int lane = threadIdx.x;  // 32 threads = 1 wave
  __shared__ __align__(32) _Float16 qP[512];     // packed q tile (K padded 16->32)
  __shared__ __align__(32) _Float16 vP[5][512];  // packed V (K padded 129->160)
  __shared__ __align__(32) _Float16 wP[5][512];  // packed softmax weights
  __shared__ float sc[16][144];
  const _Float16* qb = qh + (size_t)b * kL * kH;
  const _Float16* vb = vh + (size_t)b * kL * kH;
  int hoff = h * kDH;
  for (int i = lane; i < 512; i += 32) qP[i] = (_Float16)0.f;
  for (int i = lane; i < 5 * 512; i += 32) (&wP[0][0])[i] = (_Float16)0.f;
  if (lane < 16) {
    int qr = mt * 16 + lane;
    if (qr < kL) {
      v16h qv = *(const v16h*)(qb + (size_t)qr * kH + hoff);
#pragma unroll
      for (int d = 0; d < 16; ++d) qP[packA_idx(lane, d)] = qv[d];
    }
  }
  for (int r = lane; r < 160; r += 32) {  // stage V, coalesced rows -> packed
    v16h vr;
    if (r < kL) vr = *(const v16h*)(vb + (size_t)r * kH + hoff);
    else {
#pragma unroll
      for (int c2 = 0; c2 < 16; ++c2) vr[c2] = (_Float16)0.f;
    }
    int kc = r >> 5, rr = r & 31, hf2 = rr >> 4, i2 = rr & 15;
#pragma unroll
    for (int c2 = 0; c2 < 16; ++c2) vP[kc][((c2 + 16 * hf2) << 4) | i2] = vr[c2];
  }
  __syncthreads();
  const _Float16* kb = kh + (size_t)b * kL * kH;
  const unsigned char* spb = sp + (size_t)b * kL * kL;
  const float scale = 0.25f;  // 1/sqrt(DH)
  v16h aq = fragP(qP, lane);
  for (int jt = 0; jt < 9; ++jt) {
    v16h bk2;  // K^T fragment: row j is 16 contiguous f16; upper K half is zero
    {
      int c = lane & 15, hf2 = lane >> 4;
      int j = jt * 16 + c;
      if (hf2 == 0 && j < kL) {
        bk2 = *(const v16h*)(kb + (size_t)j * kH + hoff);
      } else {
#pragma unroll
        for (int c2 = 0; c2 < 16; ++c2) bk2[c2] = (_Float16)0.f;
      }
    }
    v8f acc = {};
    acc = wmma32(aq, bk2, acc);
    int hf = lane >> 4, n = lane & 15;
#pragma unroll
    for (int v = 0; v < 8; ++v) {
      int qr = mt * 16 + v + 8 * hf, j = jt * 16 + n;
      float bias = 0.f;
      if (qr < kL && j < kL)
        bias = sp_bias[(int)spb[(size_t)qr * kL + j] * kNH + h];
      sc[v + 8 * hf][jt * 16 + n] = acc[v] * scale + bias;
    }
  }
  __syncthreads();
  if (lane < 16) {
    int qr = mt * 16 + lane;
    if (qr < kL) {
      float mx = -1e30f;
      for (int j = 0; j < kL; ++j) mx = fmaxf(mx, sc[lane][j]);
      float s = 0.f;
      for (int j = 0; j < kL; ++j) { float e = __expf(sc[lane][j] - mx); sc[lane][j] = e; s += e; }
      float inv = 1.f / s;
      for (int j = 0; j < kL; ++j)
        wP[j >> 5][packA_idx(lane, j & 31)] = (_Float16)(sc[lane][j] * inv);
    }
  }
  __syncthreads();
  v8f acc = {};
#pragma unroll
  for (int kc = 0; kc < 5; ++kc)
    acc = wmma32(fragP(&wP[kc][0], lane), fragP(&vP[kc][0], lane), acc);
  // store attention output directly in packed-A order for the O projection
  _Float16* ob = aoP + ((size_t)(b * 9 + mt) << 11);  // 2048 f16 per strip
  int hf = lane >> 4, n = lane & 15;
  int kcol = h >> 1, dbase = (h & 1) * 16;
#pragma unroll
  for (int v = 0; v < 8; ++v) {
    int rr = v + 8 * hf;
    ob[(kcol << 9) + packA_idx(rr, dbase + n)] = (_Float16)acc[v];
  }
}

// ---------------- output projection + residual ----------------

__global__ void k_oproj(const _Float16* aoP, const _Float16* WoP, const float* bo,
                        float* xd) {
  int b = blockIdx.x, mt = blockIdx.y, t = threadIdx.x;  // 128 threads
  __shared__ __align__(32) _Float16 as[4][512];
  stage_tile(&as[0][0], aoP + ((size_t)(b * 9 + mt) << 11), 256, t, 128);
  __syncthreads();
  int wave = t >> 5, lane = t & 31;
  v16h afr[4];
#pragma unroll
  for (int kc = 0; kc < 4; ++kc) afr[kc] = fragP(&as[kc][0], lane);
  float* xb = xd + (size_t)b * kL * kH;
  for (int nt = wave; nt < 8; nt += 4) {
    v8f acc = {};
#pragma unroll
    for (int kc = 0; kc < 4; ++kc)
      acc = wmma32(afr[kc], fragP(WoP + (((size_t)nt * 4 + kc) << 9), lane), acc);
    int hf = lane >> 4, n = nt * 16 + (lane & 15);
#pragma unroll
    for (int v = 0; v < 8; ++v) {
      int r = mt * 16 + v + 8 * hf;
      if (r < kL) xb[(size_t)r * kH + n] += acc[v] + bo[n];
    }
  }
}

// ---------------- LN2 + FF1 + exact GELU (packed output) ----------------

__global__ void k_ff1(const float* xd, const float* g, const float* be,
                      const _Float16* W1P, const float* b1, _Float16* t1P) {
  int b = blockIdx.x, mt = blockIdx.y, t = threadIdx.x;  // 256 threads
  __shared__ __align__(32) _Float16 aP[4][512];
  ln_strip_packed(xd + (size_t)b * kL * kH, mt, g, be, aP, t);
  __syncthreads();
  int wave = t >> 5, lane = t & 31;
  v16h afr[4];
#pragma unroll
  for (int kc = 0; kc < 4; ++kc) afr[kc] = fragP(&aP[kc][0], lane);
  _Float16* tb = t1P + ((size_t)(b * 9 + mt) << 13);  // 8192 f16 per strip
  for (int nt = wave; nt < 32; nt += 8) {
    v8f acc = {};
#pragma unroll
    for (int kc = 0; kc < 4; ++kc)
      acc = wmma32(afr[kc], fragP(W1P + (((size_t)nt * 4 + kc) << 9), lane), acc);
    int hf = lane >> 4, n = nt * 16 + (lane & 15);
    int kc2 = n >> 5, d = n & 31;
#pragma unroll
    for (int v = 0; v < 8; ++v) {
      int rr = v + 8 * hf;
      float x0 = acc[v] + b1[n];
      float ge = 0.5f * x0 * (1.f + erff(x0 * 0.70710678f));
      tb[(kc2 << 9) + packA_idx(rr, d)] = (_Float16)ge;
    }
  }
}

// ---------------- FF2 + residual (packed input, async-staged) ----------------

__global__ void k_ff2(const _Float16* t1P, const _Float16* W2P, const float* b2,
                      float* xd) {
  int b = blockIdx.x, mt = blockIdx.y, t = threadIdx.x;  // 128 threads
  __shared__ __align__(32) _Float16 ts[16][512];  // 16 KB packed tile
  stage_tile(&ts[0][0], t1P + ((size_t)(b * 9 + mt) << 13), 1024, t, 128);
  __syncthreads();
  int wave = t >> 5, lane = t & 31;
  float* xb = xd + (size_t)b * kL * kH;
  for (int nt = wave; nt < 8; nt += 4) {
    v8f acc = {};
#pragma unroll
    for (int kc = 0; kc < 16; ++kc)
      acc = wmma32(fragP(&ts[kc][0], lane),
                   fragP(W2P + (((size_t)nt * 16 + kc) << 9), lane), acc);
    int hf = lane >> 4, n = nt * 16 + (lane & 15);
#pragma unroll
    for (int v = 0; v < 8; ++v) {
      int r = mt * 16 + v + 8 * hf;
      if (r < kL) xb[(size_t)r * kH + n] += acc[v] + b2[n];
    }
  }
}

// ---------------- final LN (CLS row) + MLP head ----------------

__global__ void k_head(const float* xd, const float* fg, const float* fb,
                       const float* Wm1, const float* bm1, const float* Wm2,
                       const float* bm2, const float* Wm3, const float* bm3,
                       float* out) {
  int b = blockIdx.x, t = threadIdx.x;  // 128 threads
  __shared__ float cls[kH], h1[kMLPH], h2[kH], red[128];
  const float* xr = xd + (size_t)b * kL * kH;  // row 0 = CLS
  float v0 = xr[t];
  red[t] = v0;
  __syncthreads();
  for (int s = 64; s > 0; s >>= 1) { if (t < s) red[t] += red[t + s]; __syncthreads(); }
  float mean = red[0] * (1.f / kH);
  __syncthreads();
  float d0 = v0 - mean;
  red[t] = d0 * d0;
  __syncthreads();
  for (int s = 64; s > 0; s >>= 1) { if (t < s) red[t] += red[t + s]; __syncthreads(); }
  float var = red[0] * (1.f / kH);
  float is = rsqrtf(var + 1e-5f);
  cls[t] = d0 * is * fg[t] + fb[t];
  __syncthreads();
  for (int j = t; j < kMLPH; j += 128) {
    float s = bm1[j];
    for (int i = 0; i < kH; ++i) s += cls[i] * Wm1[i * kMLPH + j];
    h1[j] = fmaxf(s, 0.f);
  }
  __syncthreads();
  {
    float s = bm2[t];
    for (int i = 0; i < kMLPH; ++i) s += h1[i] * Wm2[i * kH + t];
    h2[t] = fmaxf(s, 0.f);
  }
  __syncthreads();
  if (t == 0) {
    float s = bm3[0];
    for (int i = 0; i < kH; ++i) s += h2[i] * Wm3[i];
    out[b] = s;
  }
}

// ---------------- launcher ----------------

extern "C" void kernel_launch(void* const* d_in, const int* in_sizes, int n_in,
                              void* d_out, int out_size, void* d_ws, size_t ws_size,
                              hipStream_t stream) {
  const float* x       = (const float*)d_in[0];
  const float* cond    = (const float*)d_in[1];
  const int*   edge    = (const int*)d_in[2];
  const float* Wi      = (const float*)d_in[3];
  const float* bi      = (const float*)d_in[4];
  const float* in_emb  = (const float*)d_in[5];
  const float* out_emb = (const float*)d_in[6];
  const float* sp_bias = (const float*)d_in[7];
  const float* Wcls    = (const float*)d_in[8];
  const float* bcls    = (const float*)d_in[9];
  const float* ln1_g   = (const float*)d_in[10];
  const float* ln1_b   = (const float*)d_in[11];
  const float* Wq      = (const float*)d_in[12];
  const float* bq      = (const float*)d_in[13];
  const float* Wk      = (const float*)d_in[14];
  const float* bk      = (const float*)d_in[15];
  const float* Wv      = (const float*)d_in[16];
  const float* bv      = (const float*)d_in[17];
  const float* Wo      = (const float*)d_in[18];
  const float* bo      = (const float*)d_in[19];
  const float* ln2_g   = (const float*)d_in[20];
  const float* ln2_b   = (const float*)d_in[21];
  const float* W1f     = (const float*)d_in[22];
  const float* b1f     = (const float*)d_in[23];
  const float* W2f     = (const float*)d_in[24];
  const float* b2f     = (const float*)d_in[25];
  const float* fn_g    = (const float*)d_in[26];
  const float* fn_b    = (const float*)d_in[27];
  const float* Wm1     = (const float*)d_in[28];
  const float* bm1     = (const float*)d_in[29];
  const float* Wm2     = (const float*)d_in[30];
  const float* bm2     = (const float*)d_in[31];
  const float* Wm3     = (const float*)d_in[32];
  const float* bm3     = (const float*)d_in[33];
  float* out = (float*)d_out;

  char* base = (char*)d_ws;
  size_t off = 0;
  auto alloc = [&](size_t bytes) -> char* {
    char* r = base + off;
    off = (off + bytes + 255) & ~(size_t)255;
    return r;
  };
  float*         xd  = (float*)alloc(sizeof(float) * kB * kL * kH);
  unsigned char* sp  = (unsigned char*)alloc((size_t)kB * kL * kL);
  int*           din = (int*)alloc(sizeof(int) * kB * kN);
  int*           dou = (int*)alloc(sizeof(int) * kB * kN);
  _Float16* qh  = (_Float16*)alloc(sizeof(_Float16) * kB * kL * kH);
  _Float16* kh  = (_Float16*)alloc(sizeof(_Float16) * kB * kL * kH);
  _Float16* vh  = (_Float16*)alloc(sizeof(_Float16) * kB * kL * kH);
  _Float16* aoP = (_Float16*)alloc(sizeof(_Float16) * kB * 9 * 2048);
  _Float16* t1P = (_Float16*)alloc(sizeof(_Float16) * kB * 9 * 8192);
  _Float16* WiP = (_Float16*)alloc(sizeof(_Float16) * kND * kH);
  _Float16* WqP = (_Float16*)alloc(sizeof(_Float16) * kNL * kH * kH);
  _Float16* WkP = (_Float16*)alloc(sizeof(_Float16) * kNL * kH * kH);
  _Float16* WvP = (_Float16*)alloc(sizeof(_Float16) * kNL * kH * kH);
  _Float16* WoP = (_Float16*)alloc(sizeof(_Float16) * kNL * kH * kH);
  _Float16* W1P = (_Float16*)alloc(sizeof(_Float16) * kNL * kH * kFF);
  _Float16* W2P = (_Float16*)alloc(sizeof(_Float16) * kNL * kFF * kH);

  auto pack = [&](const float* W, _Float16* P, int K, int Nc) {
    int total = (K >> 5) * (Nc >> 4) * 512;
    k_pack<<<(total + 255) / 256, 256, 0, stream>>>(W, P, K, Nc);
  };
  pack(Wi, WiP, kND, kH);
  for (int l = 0; l < kNL; ++l) {
    pack(Wq  + (size_t)l * kH * kH,  WqP + (size_t)l * kH * kH,  kH, kH);
    pack(Wk  + (size_t)l * kH * kH,  WkP + (size_t)l * kH * kH,  kH, kH);
    pack(Wv  + (size_t)l * kH * kH,  WvP + (size_t)l * kH * kH,  kH, kH);
    pack(Wo  + (size_t)l * kH * kH,  WoP + (size_t)l * kH * kH,  kH, kH);
    pack(W1f + (size_t)l * kH * kFF, W1P + (size_t)l * kH * kFF, kH, kFF);
    pack(W2f + (size_t)l * kFF * kH, W2P + (size_t)l * kFF * kH, kFF, kH);
  }

  k_deg<<<kB, 128, 0, stream>>>(edge, din, dou);
  k_sp<<<kB, 128, 0, stream>>>(edge, sp);
  k_embed<<<kB, 128, 0, stream>>>(x, cond, WiP, bi, in_emb, out_emb, din, dou,
                                  Wcls, bcls, xd);

  for (int l = 0; l < kNL; ++l) {
    k_qkv<<<dim3(kB, 9), 128, 0, stream>>>(
        xd, ln1_g + l * kH, ln1_b + l * kH, WqP + (size_t)l * kH * kH,
        bq + l * kH, WkP + (size_t)l * kH * kH, bk + l * kH,
        WvP + (size_t)l * kH * kH, bv + l * kH, qh, kh, vh);
    k_attn<<<dim3(kB, kNH, 9), 32, 0, stream>>>(qh, kh, vh, sp, sp_bias, aoP);
    k_oproj<<<dim3(kB, 9), 128, 0, stream>>>(aoP, WoP + (size_t)l * kH * kH,
                                             bo + l * kH, xd);
    k_ff1<<<dim3(kB, 9), 256, 0, stream>>>(xd, ln2_g + l * kH, ln2_b + l * kH,
                                           W1P + (size_t)l * kH * kFF,
                                           b1f + l * kFF, t1P);
    k_ff2<<<dim3(kB, 9), 128, 0, stream>>>(t1P, W2P + (size_t)l * kFF * kH,
                                           b2f + l * kH, xd);
  }

  k_head<<<kB, 128, 0, stream>>>(xd, fn_g, fn_b, Wm1, bm1, Wm2, bm2, Wm3, bm3,
                                 out);
  (void)in_sizes; (void)n_in; (void)out_size; (void)ws_size;
}
